// CausalSelfAttention_71846212927979
// MI455X (gfx1250) — matrix operand order repaired and checked by reading, not compiled, not run on hardware
//
#include <hip/hip_runtime.h>
#include <hip/hip_bf16.h>

typedef __bf16 bf16_t;
typedef bf16_t v16bf __attribute__((ext_vector_type(16)));
typedef float  v8f   __attribute__((ext_vector_type(8)));
typedef unsigned int v4u __attribute__((ext_vector_type(4)));

#define TD 4096      // T == D == N == 4096
#define BM 128
#define BN 128
#define BK 64        // two v_wmma k-steps per stage
#define LDK 72       // padded LDS K-stride: 144 B rows -> conflict-free, 16B-aligned
#define NSTAGE (TD / BK)

static __device__ __forceinline__ unsigned short f32_to_bf16_rne(float f) {
  unsigned int u = __builtin_bit_cast(unsigned int, f);
  unsigned int lsb = (u >> 16) & 1u;
  u += 0x7fffu + lsb;                       // round to nearest even
  return (unsigned short)(u >> 16);
}

static __device__ __forceinline__ unsigned pack_bf16x2(float a, float b) {
  return (unsigned)f32_to_bf16_rne(a) | ((unsigned)f32_to_bf16_rne(b) << 16);
}

// Low 32 bits of a generic pointer to __shared__ = LDS byte offset (aperture rule).
static __device__ __forceinline__ unsigned lds_addr(const void* p) {
  return (unsigned)(unsigned long long)p;
}

// CDNA5 async global->LDS copy, 16 B per lane, tracked with ASYNCcnt.
static __device__ __forceinline__ void async_b128(unsigned lds, const void* sbase,
                                                  unsigned voff) {
  asm volatile("global_load_async_to_lds_b128 %0, %1, %2"
               :
               : "v"(lds), "v"(voff), "s"(sbase)
               : "memory");
}

static __device__ __forceinline__ void wait_async_le8() {
#if __has_builtin(__builtin_amdgcn_s_wait_asynccnt)
  __builtin_amdgcn_s_wait_asynccnt(8);
#else
  asm volatile("s_wait_asynccnt 8" ::: "memory");
#endif
}
static __device__ __forceinline__ void wait_async_le0() {
#if __has_builtin(__builtin_amdgcn_s_wait_asynccnt)
  __builtin_amdgcn_s_wait_asynccnt(0);
#else
  asm volatile("s_wait_asynccnt 0" ::: "memory");
#endif
}

// Build a v16bf WMMA fragment from two 16-byte LDS chunks (ISA §7.12.2 layouts).
struct U8x2 { v4u lo, hi; };
static __device__ __forceinline__ v16bf make_frag(const unsigned short* p0,
                                                  const unsigned short* p1) {
  U8x2 u{*(const v4u*)p0, *(const v4u*)p1};
  return __builtin_bit_cast(v16bf, u);
}

// ---------------------------------------------------------------------------
// C[M,N] = A_bf16[M,K] * Bt_bf16[N,K]^T   (all five GEMMs use this form)
//   BF16OUT=true : C bf16, += bias[N]                (Q/K/V projections)
//   BF16OUT=false: C f32,  *= scale                  (scores, P*V^T)
// Double-buffered LDS, staged with global_load_async_to_lds_b128.
// ---------------------------------------------------------------------------
template <bool BF16OUT>
__global__ __launch_bounds__(256) void gemm_nt(
    const unsigned short* __restrict__ A, const unsigned short* __restrict__ Bt,
    void* __restrict__ Cout, const float* __restrict__ bias, float scale) {
  __shared__ unsigned short As[2][BM][LDK];
  __shared__ unsigned short Bs[2][BN][LDK];

  const int tid = threadIdx.x;
  const int lane = tid & 31, wid = tid >> 5;
  const int wr = wid >> 1, wc = wid & 1;       // 4x2 wave grid over 128x128
  const int nl = lane & 15, hl = lane >> 4;    // lane -> (row/col, K-half)
  const int bM = blockIdx.y * BM, bN = blockIdx.x * BN;

  // Staging: each thread copies 32 bf16 (4 x b128) per tile per stage.
  const int sr = tid >> 1;                     // tile row 0..127
  const int sk = (tid & 1) * 32;               // k start 0 / 32
  const unsigned aoff0 = (unsigned)(((bM + sr) * TD + sk) * 2);
  const unsigned boff0 = (unsigned)(((bN + sr) * TD + sk) * 2);
  const unsigned ldsA = lds_addr(&As[0][sr][sk]);
  const unsigned ldsB = lds_addr(&Bs[0][sr][sk]);
  const unsigned bufStride = (unsigned)(BM * LDK * 2);   // bytes per buffer

  v8f acc[2][4] = {};

  auto stage = [&](int buf, int s) {
    const unsigned kbyte = (unsigned)(s * BK * 2);
    const unsigned la = ldsA + (unsigned)buf * bufStride;
    const unsigned lb = ldsB + (unsigned)buf * bufStride;
#pragma unroll
    for (int c = 0; c < 4; ++c)
      async_b128(la + c * 16, A, aoff0 + kbyte + c * 16);
#pragma unroll
    for (int c = 0; c < 4; ++c)
      async_b128(lb + c * 16, Bt, boff0 + kbyte + c * 16);
  };

  stage(0, 0);
  for (int s = 0; s < NSTAGE; ++s) {
    const int buf = s & 1;
    if (s + 1 < NSTAGE) { stage(buf ^ 1, s + 1); wait_async_le8(); }
    else                { wait_async_le0(); }
    __syncthreads();                            // stage s visible to all waves

#pragma unroll
    for (int kk = 0; kk < BK; kk += 32) {
      v16bf afr[2], bfr[4];
#pragma unroll
      for (int ri = 0; ri < 2; ++ri) {
        int m = wr * 32 + ri * 16 + nl;
        afr[ri] = make_frag(&As[buf][m][kk + hl * 8], &As[buf][m][kk + 16 + hl * 8]);
      }
#pragma unroll
      for (int ci = 0; ci < 4; ++ci) {
        int n = wc * 64 + ci * 16 + nl;
        bfr[ci] = make_frag(&Bs[buf][n][kk + hl * 8], &Bs[buf][n][kk + 16 + hl * 8]);
      }
#pragma unroll
      for (int ri = 0; ri < 2; ++ri)
#pragma unroll
        for (int ci = 0; ci < 4; ++ci)
          acc[ri][ci] = __builtin_amdgcn_wmma_f32_16x16x32_bf16(
              false, afr[ri], false, bfr[ci], (short)0, acc[ri][ci], false, false);
    }
    __syncthreads();                            // done reading buf before re-fill
  }

#pragma unroll
  for (int ci = 0; ci < 4; ++ci) {
    int n = bN + wc * 64 + ci * 16 + nl;
    float bvv = BF16OUT ? bias[n] : 0.0f;
#pragma unroll
    for (int ri = 0; ri < 2; ++ri) {
      int mbase = bM + wr * 32 + ri * 16 + hl * 8;
#pragma unroll
      for (int v = 0; v < 8; ++v) {
        size_t idx = (size_t)(mbase + v) * TD + n;
        if (BF16OUT)
          ((unsigned short*)Cout)[idx] = f32_to_bf16_rne(acc[ri][ci][v] + bvv);
        else
          ((float*)Cout)[idx] = acc[ri][ci][v] * scale;
      }
    }
  }
}

// ---------------------------------------------------------------------------
// Elementwise fp32 -> bf16 (x), packed 8 elems / thread.
// ---------------------------------------------------------------------------
__global__ __launch_bounds__(256) void cast_f32_bf16(
    const float* __restrict__ in, unsigned short* __restrict__ out) {
  size_t i = ((size_t)blockIdx.x * 256 + threadIdx.x) * 8;
  float4 f0 = *(const float4*)(in + i);
  float4 f1 = *(const float4*)(in + i + 4);
  v4u o;
  o.x = pack_bf16x2(f0.x, f0.y);
  o.y = pack_bf16x2(f0.z, f0.w);
  o.z = pack_bf16x2(f1.x, f1.y);
  o.w = pack_bf16x2(f1.z, f1.w);
  *(v4u*)(out + i) = o;
}

// ---------------------------------------------------------------------------
// Wt_bf16[n][k] = W_f32[k][n]   (64x64 LDS tile transpose, coalesced both sides)
// ---------------------------------------------------------------------------
__global__ __launch_bounds__(256) void transpose_f32_bf16(
    const float* __restrict__ in, unsigned short* __restrict__ out) {
  __shared__ unsigned short t[64][72];
  const int n0 = blockIdx.x * 64, k0 = blockIdx.y * 64;
  const int tx = threadIdx.x & 63, ty = threadIdx.x >> 6;   // 64 x 4
#pragma unroll
  for (int r = 0; r < 64; r += 4)
    t[ty + r][tx] = f32_to_bf16_rne(in[(size_t)(k0 + ty + r) * TD + n0 + tx]);
  __syncthreads();
#pragma unroll
  for (int r = 0; r < 64; r += 4)
    out[(size_t)(n0 + ty + r) * TD + k0 + tx] = t[tx][ty + r];
}

// ---------------------------------------------------------------------------
// Row softmax: one block per row; P_bf16 = softmax(S_f32) row-wise.
// ---------------------------------------------------------------------------
__global__ __launch_bounds__(256) void softmax_rows(
    const float* __restrict__ S, unsigned short* __restrict__ P) {
  __shared__ float red[256];
  const int row = blockIdx.x, tid = threadIdx.x;
  const float* s = S + (size_t)row * TD;

  float vals[16];
  float m = -3.4e38f;
#pragma unroll
  for (int i = 0; i < 16; ++i) { vals[i] = s[tid + i * 256]; m = fmaxf(m, vals[i]); }
  red[tid] = m; __syncthreads();
  for (int off = 128; off > 0; off >>= 1) {
    if (tid < off) red[tid] = fmaxf(red[tid], red[tid + off]);
    __syncthreads();
  }
  m = red[0]; __syncthreads();

  float l = 0.0f;
#pragma unroll
  for (int i = 0; i < 16; ++i) { vals[i] = __expf(vals[i] - m); l += vals[i]; }
  red[tid] = l; __syncthreads();
  for (int off = 128; off > 0; off >>= 1) {
    if (tid < off) red[tid] += red[tid + off];
    __syncthreads();
  }
  float inv = 1.0f / red[0];
#pragma unroll
  for (int i = 0; i < 16; ++i)
    P[(size_t)row * TD + tid + i * 256] = f32_to_bf16_rne(vals[i] * inv);
}

// ---------------------------------------------------------------------------
extern "C" void kernel_launch(void* const* d_in, const int* in_sizes, int n_in,
                              void* d_out, int out_size, void* d_ws, size_t ws_size,
                              hipStream_t stream) {
  (void)in_sizes; (void)n_in; (void)out_size; (void)ws_size;
  const float* x  = (const float*)d_in[0];
  const float* Wq = (const float*)d_in[1];
  const float* bq = (const float*)d_in[2];
  const float* Wk = (const float*)d_in[3];
  const float* bk = (const float*)d_in[4];
  const float* Wv = (const float*)d_in[5];
  const float* bv = (const float*)d_in[6];
  float* out = (float*)d_out;

  const size_t E = (size_t)TD * TD;
  unsigned short* xb  = (unsigned short*)d_ws;  // bf16 x
  unsigned short* Wqt = xb + E;                 // bf16 Wq^T
  unsigned short* Wkt = xb + 2 * E;             // bf16 Wk^T
  unsigned short* Wvt = xb + 3 * E;             // bf16 Wv^T
  unsigned short* Qb  = xb + 4 * E;             // bf16 Q
  unsigned short* Kb  = xb + 5 * E;             // bf16 K
  unsigned short* Vb  = xb + 6 * E;             // bf16 V
  float*          S   = (float*)Wqt;            // fp32 scores (Wq^T/Wk^T dead by then)
  unsigned short* Pb  = Wvt;                    // bf16 P (Wv^T dead by then)

  dim3 gG(TD / BN, TD / BM), gT(TD / 64, TD / 64), blk(256);
  cast_f32_bf16<<<E / (256 * 8), blk, 0, stream>>>(x, xb);
  transpose_f32_bf16<<<gT, blk, 0, stream>>>(Wq, Wqt);
  transpose_f32_bf16<<<gT, blk, 0, stream>>>(Wk, Wkt);
  transpose_f32_bf16<<<gT, blk, 0, stream>>>(Wv, Wvt);
  gemm_nt<true ><<<gG, blk, 0, stream>>>(xb, Wqt, Qb, bq, 1.0f);
  gemm_nt<true ><<<gG, blk, 0, stream>>>(xb, Wkt, Kb, bk, 1.0f);
  gemm_nt<true ><<<gG, blk, 0, stream>>>(xb, Wvt, Vb, bv, 1.0f);
  gemm_nt<false><<<gG, blk, 0, stream>>>(Qb, Kb, S, nullptr, 1.0f / 64.0f);
  softmax_rows<<<TD, blk, 0, stream>>>(S, Pb);
  gemm_nt<false><<<gG, blk, 0, stream>>>(Pb, Vb, out, nullptr, 1.0f);
}
